// Net_22101901705285
// MI455X (gfx1250) — matrix operand order repaired and checked by reading, not compiled, not run on hardware
//
#include <hip/hip_runtime.h>
#include <hip/hip_bf16.h>

typedef __attribute__((ext_vector_type(16))) _Float16 v16h;
typedef __attribute__((ext_vector_type(8)))  float    v8f;
typedef __attribute__((ext_vector_type(4)))  float    v4f;

#define TK 32   // K-chunk per WMMA

// ---------------------------------------------------------------------------
// WMMA GEMM: Y[M x Nf] = A[M x K] @ B[K x Nf]; f32 in/out, f16 multiply,
// f32 accumulate (v_wmma_f32_16x16x32_f16).
//
// Block = 128 threads (4 waves), 64x64 output tile:
//   wave w -> M-subtile w; 4 accumulators per wave (4 N-subtiles) ->
//   4 WMMAs per wave per K-chunk, A-fragment reused, B-fragments shared.
//
// LDS holds tiles in fragment order (lane's 16 halves contiguous, 32B
// aligned): fragment load = 2x ds_load_b128. Staging: thread t owns slice
// (sub = t>>5, l = t&31); guards are clamp+select (branch-free); A staged
// with 4x global_load_b128 on the K%32==0 fast path.
//
// Fragment element mapping (ISA 7.12.2, f16 16x16x32):
//   A: half=lane>>4, row=lane&15; a[j] = A[row][half*8 + j + (j>=8)*8]
//   B: col=lane&15;               b[j] = B[half*8 + j + (j>=8)*8][col]
//   D: VGPR v -> row = 16*wave + (half?8:0) + v; col = lane&15
// ---------------------------------------------------------------------------
__global__ void gemm_wmma_f16(const float* __restrict__ A,
                              const float* __restrict__ B,
                              float* __restrict__ Y,
                              int M, int K, int Nf) {
  __shared__ alignas(32) _Float16 sA[4 * 32 * 16];  // [mSub][lane][j]
  __shared__ alignas(32) _Float16 sB[4 * 32 * 16];  // [nSub][lane][j]

  const int tid  = threadIdx.x;
  const int wave = tid >> 5;
  const int lane = tid & 31;
  const int mBase = blockIdx.x * 64;
  const int nBase = blockIdx.y * 64;

  // staging slice owned by this thread
  const int half  = lane >> 4;                       // 0 | 1
  const int aRow  = mBase + wave * 16 + (lane & 15); // row for A slice
  const int aRowC = min(aRow, M - 1);                // clamped (in-bounds)
  const _Float16 aMask = (aRow < M) ? (_Float16)1.0f : (_Float16)0.0f;
  const int bCol  = nBase + wave * 16 + (lane & 15); // col for B slice
  const float* __restrict__ aPtr = A + (size_t)aRowC * K;

  v8f acc0 = {}, acc1 = {}, acc2 = {}, acc3 = {};

  for (int k0 = 0; k0 < K; k0 += TK) {
    // ---- stage A slice: 16 halves, two contiguous 8-float runs ----
    {
      const int kBase = k0 + half * 8;     // runs: [kBase,+8) and [kBase+16,+8)
      v16h fa;
      if (kBase + 24 <= K) {               // fast path (K % 32 == 0 layers)
        v4f p0 = *(const v4f*)(aPtr + kBase);
        v4f p1 = *(const v4f*)(aPtr + kBase + 4);
        v4f p2 = *(const v4f*)(aPtr + kBase + 16);
        v4f p3 = *(const v4f*)(aPtr + kBase + 20);
#pragma unroll
        for (int j = 0; j < 4; ++j) {
          fa[j]      = (_Float16)p0[j];
          fa[4 + j]  = (_Float16)p1[j];
          fa[8 + j]  = (_Float16)p2[j];
          fa[12 + j] = (_Float16)p3[j];
        }
      } else {                             // guarded path (layer 1, K = 8)
#pragma unroll
        for (int j = 0; j < 16; ++j) {
          int k   = kBase + ((j < 8) ? j : (j + 8));
          float v = aPtr[min(k, K - 1)];
          fa[j]   = (_Float16)((k < K) ? v : 0.0f);
        }
      }
      fa = fa * aMask;                     // zero rows >= M
      ((v16h*)sA)[wave * 32 + lane] = fa;  // 2x ds_store_b128
    }
    // ---- stage B slice: 16 halves, column bCol, rows [kBase,+8) and [kBase+16,+8) ----
    {
      const int kBase = k0 + half * 8;
      v16h fb;
      if (kBase + 24 <= K) {               // fast path
#pragma unroll
        for (int j = 0; j < 16; ++j)
          fb[j] = (_Float16)B[(size_t)(kBase + ((j < 8) ? j : (j + 8))) * Nf + bCol];
      } else {                             // guarded path (layer 1)
#pragma unroll
        for (int j = 0; j < 16; ++j) {
          int k   = kBase + ((j < 8) ? j : (j + 8));
          float v = B[(size_t)min(k, K - 1) * Nf + bCol];
          fb[j]   = (_Float16)((k < K) ? v : 0.0f);
        }
      }
      ((v16h*)sB)[wave * 32 + lane] = fb;
    }
    __syncthreads();

    const v16h a  = ((const v16h*)sA)[wave * 32 + lane];   // 2x ds_load_b128
    const v16h b0 = ((const v16h*)sB)[0 * 32 + lane];
    const v16h b1 = ((const v16h*)sB)[1 * 32 + lane];
    const v16h b2 = ((const v16h*)sB)[2 * 32 + lane];
    const v16h b3 = ((const v16h*)sB)[3 * 32 + lane];

    acc0 = __builtin_amdgcn_wmma_f32_16x16x32_f16(false, a, false, b0,
                                                  (short)0, acc0, false, false);
    acc1 = __builtin_amdgcn_wmma_f32_16x16x32_f16(false, a, false, b1,
                                                  (short)0, acc1, false, false);
    acc2 = __builtin_amdgcn_wmma_f32_16x16x32_f16(false, a, false, b2,
                                                  (short)0, acc2, false, false);
    acc3 = __builtin_amdgcn_wmma_f32_16x16x32_f16(false, a, false, b3,
                                                  (short)0, acc3, false, false);
    __syncthreads();
  }

  const int rowBase = mBase + wave * 16 + ((lane >> 4) ? 8 : 0);
  const int colBase = nBase + (lane & 15);
  if (rowBase + 7 < M) {                   // fast path: whole subtile in range
#pragma unroll
    for (int v = 0; v < 8; ++v) {
      size_t o = (size_t)(rowBase + v) * Nf + colBase;
      Y[o +  0] = acc0[v];
      Y[o + 16] = acc1[v];
      Y[o + 32] = acc2[v];
      Y[o + 48] = acc3[v];
    }
  } else {
#pragma unroll
    for (int v = 0; v < 8; ++v) {
      int row = rowBase + v;
      if (row < M) {
        size_t o = (size_t)row * Nf + colBase;
        Y[o +  0] = acc0[v];
        Y[o + 16] = acc1[v];
        Y[o + 32] = acc2[v];
        Y[o + 48] = acc3[v];
      }
    }
  }
}

// ---------------------------------------------------------------------------
// Degree / normalization
// ---------------------------------------------------------------------------
__global__ void deg_init(float* deg, int n) {
  int i = blockIdx.x * blockDim.x + threadIdx.x;
  if (i < n) deg[i] = 1.0f;                 // self-loop weight
}
__global__ void deg_edges(const int* __restrict__ dst,
                          const float* __restrict__ ew, float* deg, int e) {
  int i = blockIdx.x * blockDim.x + threadIdx.x;
  if (i < e) atomicAdd(&deg[dst[i]], ew[i]);
}
__global__ void deg_to_dinv(float* deg, int n) {
  int i = blockIdx.x * blockDim.x + threadIdx.x;
  if (i < n) { float d = deg[i]; deg[i] = (d > 0.0f) ? rsqrtf(d) : 0.0f; }
}
__global__ void edge_norm(const int* __restrict__ src, const int* __restrict__ dst,
                          const float* __restrict__ ew,
                          const float* __restrict__ dinv,
                          float* __restrict__ nrm, int e) {
  int i = blockIdx.x * blockDim.x + threadIdx.x;
  if (i < e) nrm[i] = dinv[src[i]] * ew[i] * dinv[dst[i]];
}

// ---------------------------------------------------------------------------
// Message passing: AGG[dst] += Y[src] * norm   (grid: (E, F/128), block 128)
// ---------------------------------------------------------------------------
__global__ void edge_scatter(const float* __restrict__ Y,
                             const int* __restrict__ src,
                             const int* __restrict__ dst,
                             const float* __restrict__ nrm,
                             float* __restrict__ AGG, int F) {
  const int e = blockIdx.x;
  const int f = blockIdx.y * 128 + threadIdx.x;
  const float nv = nrm[e];
  const int s = src[e], d = dst[e];
  atomicAdd(&AGG[(size_t)d * F + f], Y[(size_t)s * F + f] * nv);
}

// Self-loop + bias + ReLU:  H = relu(AGG + Y*dinv^2 + b)
__global__ void bias_relu_selfloop(const float* __restrict__ AGG,
                                   const float* __restrict__ Y,
                                   const float* __restrict__ dinv,
                                   const float* __restrict__ b,
                                   float* __restrict__ H, int F) {
  const int i = blockIdx.x;
  const int f = blockIdx.y * 128 + threadIdx.x;
  const float di = dinv[i];
  const float v = AGG[(size_t)i * F + f] + Y[(size_t)i * F + f] * di * di + b[f];
  H[(size_t)i * F + f] = fmaxf(v, 0.0f);
}

// ---------------------------------------------------------------------------
// Global mean pool + FC + sigmoid
// ---------------------------------------------------------------------------
__global__ void pool_scatter(const float* __restrict__ H,
                             const int* __restrict__ batch,
                             float* __restrict__ pooled,
                             float* __restrict__ cnt, int F) {
  const int i = blockIdx.x;
  const int f = blockIdx.y * 128 + threadIdx.x;
  const int g = batch[i];
  atomicAdd(&pooled[(size_t)g * F + f], H[(size_t)i * F + f]);
  if (f == 0) atomicAdd(&cnt[g], 1.0f);
}

__global__ void pool_finalize(const float* __restrict__ pooled,
                              const float* __restrict__ cnt,
                              const float* __restrict__ Wfc,
                              const float* __restrict__ bfc,
                              float* __restrict__ out, int G, int F) {
  const int g = blockIdx.x * blockDim.x + threadIdx.x;
  if (g >= G) return;
  const float c = fmaxf(cnt[g], 1.0f);
  float s = 0.0f;
  for (int f = 0; f < F; ++f) s += pooled[(size_t)g * F + f] * Wfc[f];
  s = s / c + bfc[0];
  out[g] = 1.0f / (1.0f + __expf(-s));
}

// ---------------------------------------------------------------------------
extern "C" void kernel_launch(void* const* d_in, const int* in_sizes, int n_in,
                              void* d_out, int out_size, void* d_ws, size_t ws_size,
                              hipStream_t stream) {
  const int N = 50000, E = 150000, G = 256;
  const int H3 = 512;

  const float* x    = (const float*)d_in[0];
  const int*   ei   = (const int*)d_in[1];
  const int*   srcI = ei;
  const int*   dstI = ei + E;
  const float* ew   = (const float*)d_in[2];
  const int*   batch= (const int*)d_in[3];
  const float* W1   = (const float*)d_in[4];
  const float* b1   = (const float*)d_in[5];
  const float* W2   = (const float*)d_in[6];
  const float* b2   = (const float*)d_in[7];
  const float* W3   = (const float*)d_in[8];
  const float* b3   = (const float*)d_in[9];
  const float* Wfc  = (const float*)d_in[10];
  const float* bfc  = (const float*)d_in[11];

  float* ws     = (float*)d_ws;
  float* dinv   = ws;                                // N (deg -> dinv in place)
  float* nrm    = dinv + N;                          // E
  float* Y      = nrm + E;                           // N*512
  float* AGG    = Y   + (size_t)N * H3;              // N*512
  float* Hbuf   = AGG + (size_t)N * H3;              // N*512
  float* pooled = Hbuf + (size_t)N * H3;             // G*512
  float* cnt    = pooled + (size_t)G * H3;           // G

  // degree / normalization
  deg_init   <<<(N + 255) / 256, 256, 0, stream>>>(dinv, N);
  deg_edges  <<<(E + 255) / 256, 256, 0, stream>>>(dstI, ew, dinv, E);
  deg_to_dinv<<<(N + 255) / 256, 256, 0, stream>>>(dinv, N);
  edge_norm  <<<(E + 255) / 256, 256, 0, stream>>>(srcI, dstI, ew, dinv, nrm, E);

  auto run_layer = [&](const float* Hin, const float* W, const float* b,
                       int K, int F) {
    dim3 gg((N + 63) / 64, F / 64);
    gemm_wmma_f16<<<gg, 128, 0, stream>>>(Hin, W, Y, N, K, F);
    hipMemsetAsync(AGG, 0, (size_t)N * F * sizeof(float), stream);
    dim3 ge(E, F / 128);
    edge_scatter<<<ge, 128, 0, stream>>>(Y, srcI, dstI, nrm, AGG, F);
    dim3 gn(N, F / 128);
    bias_relu_selfloop<<<gn, 128, 0, stream>>>(AGG, Y, dinv, b, Hbuf, F);
  };

  run_layer(x,    W1, b1,   8, 128);   // h1 in Hbuf
  run_layer(Hbuf, W2, b2, 128, 256);   // h2 in Hbuf
  run_layer(Hbuf, W3, b3, 256, 512);   // h3 in Hbuf

  // pooling + FC + sigmoid
  hipMemsetAsync(pooled, 0, ((size_t)G * H3 + G) * sizeof(float), stream);
  pool_scatter <<<dim3(N, H3 / 128), 128, 0, stream>>>(Hbuf, batch, pooled, cnt, H3);
  pool_finalize<<<(G + 255) / 256, 256, 0, stream>>>(pooled, cnt, Wfc, bfc,
                                                     (float*)d_out, G, H3);
}
